// BidirRecurrentModel_47270410060257
// MI455X (gfx1250) — matrix operand order repaired and checked by reading, not compiled, not run on hardware
//
#include <hip/hip_runtime.h>
#include <hip/hip_bf16.h>
#include <math.h>

typedef __attribute__((ext_vector_type(16))) __bf16 v16bf;
typedef __attribute__((ext_vector_type(8)))  float  v8f;

#define T_SEQ 2048
#define I_IN  256
#define HID   256
#define NG    1024   // 4*HID
#define NWG   16
#define NTHR  256

// ---- workspace layout (bytes) ----
#define OFF_BPACK0 ((size_t)0)         // 16kt x 64nt x 512 bf16 = 1 MB
#define OFF_BPACK1 ((size_t)1048576)
#define OFF_BPACKF ((size_t)2097152)   // 16kt x 16nt x 512 bf16 = 256 KB
#define OFF_BIAS0  ((size_t)2359296)   // 1024 f32
#define OFF_BIAS1  ((size_t)2363392)
#define OFF_H0     ((size_t)2367488)   // 128x256 f32 each
#define OFF_C0     ((size_t)2498560)
#define OFF_H1     ((size_t)2629632)
#define OFF_C1     ((size_t)2760704)
#define OFF_HB1    ((size_t)2891776)
#define OFF_CB     ((size_t)3022848)   // scratch c for backward pre-pass
#define OFF_APACK0 ((size_t)3153920)   // 128x512 bf16 fragment-packed
#define OFF_APACK1 ((size_t)3284992)
#define OFF_BAR    ((size_t)3416064)   // 2 x u32

__device__ __forceinline__ float sigm(float v) { return 1.0f / (1.0f + __expf(-v)); }

// Scatter one f32 value into the WMMA A-fragment-packed bf16 buffer.
// A fragment (16x32 bf16): lane = M | (K hi-half bit << 4),
// per-lane element j covers K = j + (lane>=16 ? 16 : 0).
__device__ __forceinline__ void packA(__bf16* Ap, int b, int K, float v)
{
    const int mt   = b >> 4;
    const int kt   = K >> 5;
    const int kp   = K & 31;
    const int lane = (b & 15) | (((kp >> 4) & 1) << 4);
    const int j    = kp & 15;
    Ap[((size_t)((mt * 16 + kt) * 32 + lane) << 4) + j] = (__bf16)v;
}

// Sense/generation global barrier for the persistent kernel (NWG blocks).
__device__ __forceinline__ void grid_sync(unsigned* cnt, unsigned* gen)
{
    __threadfence();
    __syncthreads();
    if (threadIdx.x == 0) {
        unsigned g = atomicAdd(gen, 0u);
        unsigned a = atomicAdd(cnt, 1u);
        if (a == (unsigned)(NWG - 1)) {
            atomicExch(cnt, 0u);
            __threadfence();
            atomicAdd(gen, 1u);
        } else {
            while (atomicAdd(gen, 0u) == g) { __builtin_amdgcn_s_sleep(2); }
        }
    }
    __syncthreads();
    __threadfence();
}

// One fused gate-GEMM + LSTM activation. Block g owns N-tiles {g,g+16,g+32,g+48}
// (the i/f/g/o columns for hidden units n = g*16..g*16+15); wave mt owns M-tile mt.
// K-loop: 4 WMMAs per chunk, double-buffered LDS for B (shared by 8 waves),
// A-fragment and B-slice for kt+1 issued one iteration early.
// After the GEMM every lane holds i/f/g/o for its 8 rows in registers and applies
// the nonlinearity in place; c is updated in cbuf; h returned in hout[8].
__device__ __forceinline__ void gemm_act(const __bf16* __restrict__ Ap,
                                         const __bf16* __restrict__ Bp,
                                         __bf16* lds, int g, int mt, int lane, int n,
                                         float4 biasv, float* __restrict__ cbuf,
                                         bool zero_c, float* __restrict__ hout)
{
    const int tid = threadIdx.x;
    const int q   = tid >> 6;       // which of the 4 B tiles this thread stages
    const int w   = tid & 63;       // float4 slot within the 1 KB tile
    const float4* B4 = (const float4*)Bp;

    v8f A0 = {}, A1 = {}, A2 = {}, A3 = {};
    float4 breg = B4[((size_t)(g + 16 * q)) * 64 + w];
    v16bf areg  = *(const v16bf*)(Ap + ((size_t)(mt * 16) << 9) + lane * 16);

    for (int kt = 0; kt < 16; ++kt) {
        const int ktn = (kt + 1 < 16) ? kt + 1 : kt;
        float4 bnext = B4[((size_t)(ktn * 64 + g + 16 * q)) * 64 + w];
        v16bf  anext = *(const v16bf*)(Ap + ((size_t)(mt * 16 + ktn) << 9) + lane * 16);

        __bf16* lb = lds + (kt & 1) * 2048;
        ((float4*)lb)[tid] = breg;
        __syncthreads();

        v16bf b0 = *(const v16bf*)(lb + lane * 16);
        v16bf b1 = *(const v16bf*)(lb + 512  + lane * 16);
        v16bf b2 = *(const v16bf*)(lb + 1024 + lane * 16);
        v16bf b3 = *(const v16bf*)(lb + 1536 + lane * 16);
        A0 = __builtin_amdgcn_wmma_f32_16x16x32_bf16(false, areg, false, b0, (short)0, A0, false, false);
        A1 = __builtin_amdgcn_wmma_f32_16x16x32_bf16(false, areg, false, b1, (short)0, A1, false, false);
        A2 = __builtin_amdgcn_wmma_f32_16x16x32_bf16(false, areg, false, b2, (short)0, A2, false, false);
        A3 = __builtin_amdgcn_wmma_f32_16x16x32_bf16(false, areg, false, b3, (short)0, A3, false, false);

        breg = bnext;
        areg = anext;
    }

    const int Mb = mt * 16 + (lane >> 4) * 8;
    #pragma unroll
    for (int r = 0; r < 8; ++r) {
        const int idx = (Mb + r) * HID + n;
        float iv = A0[r] + biasv.x;
        float fv = A1[r] + biasv.y;
        float gv = A2[r] + biasv.z;
        float ov = A3[r] + biasv.w;
        float cp = zero_c ? 0.0f : cbuf[idx];
        float cn = cp * sigm(fv) + sigm(iv) * tanhf(gv);
        cbuf[idx] = cn;
        hout[r] = sigm(ov) * tanhf(cn);
    }
}

// ---- weight packing: Bcomb[k][n] = (k<256 ? Wx[l][n][k] : Wh[l][n][k-256]),
// packed in B-fragment order (lane = N | K-half<<4, j = K within half). ----
__global__ void pack_gate_weights(const float* __restrict__ Wxh,
                                  const float* __restrict__ Whh,
                                  __bf16* __restrict__ Bp)
{
    int gid = blockIdx.x * blockDim.x + threadIdx.x;   // 0..65535
    int layer = gid >> 15;
    int r     = gid & 32767;
    int tileIdx = r >> 5;     // 0..1023  (kt*64 + nt)
    int lane    = r & 31;
    int kt  = tileIdx >> 6;
    int ntl = tileIdx & 63;
    int hi  = (lane >> 4) & 1;
    int N   = ntl * 16 + (lane & 15);
    __bf16* dst = Bp + (size_t)layer * 16 * 64 * 512 + ((size_t)tileIdx << 9) + lane * 16;
    const float* WxL = Wxh + (size_t)layer * NG * HID;
    const float* WhL = Whh + (size_t)layer * NG * HID;
    #pragma unroll
    for (int j = 0; j < 16; ++j) {
        int K = kt * 32 + j + hi * 16;
        float v = (K < HID) ? WxL[(size_t)N * HID + K] : WhL[(size_t)N * HID + (K - HID)];
        dst[j] = (__bf16)v;
    }
}

__global__ void pack_fc_weights(const float* __restrict__ fcW, __bf16* __restrict__ Bp)
{
    int gid = blockIdx.x * blockDim.x + threadIdx.x;   // 0..8191
    int tileIdx = gid >> 5;    // 0..255  (kt*16 + nt)
    int lane    = gid & 31;
    int kt  = tileIdx >> 4;
    int ntl = tileIdx & 15;
    int hi  = (lane >> 4) & 1;
    int N   = ntl * 16 + (lane & 15);
    __bf16* dst = Bp + ((size_t)tileIdx << 9) + lane * 16;
    #pragma unroll
    for (int j = 0; j < 16; ++j) {
        int K = kt * 32 + j + hi * 16;
        dst[j] = (__bf16)fcW[(size_t)N * (2 * HID) + K];
    }
}

__global__ void pack_bias(const float* __restrict__ bxh, const float* __restrict__ bhh,
                          float* __restrict__ bias01, unsigned* __restrict__ bar)
{
    int gid = blockIdx.x * blockDim.x + threadIdx.x;
    if (gid < 2 * NG) bias01[gid] = bxh[gid] + bhh[gid];
    if (gid == 0) { bar[0] = 0u; bar[1] = 0u; }
}

__global__ void __launch_bounds__(NTHR)
lstm_persistent(const float* __restrict__ x, const float* __restrict__ fcb,
                unsigned char* __restrict__ ws, float* __restrict__ out)
{
    __shared__ __align__(16) __bf16 ldsB[4096];   // 2 x 4 KB double buffer
    const __bf16* Bp0 = (const __bf16*)(ws + OFF_BPACK0);
    const __bf16* Bp1 = (const __bf16*)(ws + OFF_BPACK1);
    const __bf16* BpF = (const __bf16*)(ws + OFF_BPACKF);
    const float* bias0 = (const float*)(ws + OFF_BIAS0);
    const float* bias1 = (const float*)(ws + OFF_BIAS1);
    float* h0  = (float*)(ws + OFF_H0);
    float* c0  = (float*)(ws + OFF_C0);
    float* h1  = (float*)(ws + OFF_H1);
    float* c1  = (float*)(ws + OFF_C1);
    float* hb1 = (float*)(ws + OFF_HB1);
    float* cb  = (float*)(ws + OFF_CB);
    __bf16* Ap0 = (__bf16*)(ws + OFF_APACK0);
    __bf16* Ap1 = (__bf16*)(ws + OFF_APACK1);
    unsigned* bar = (unsigned*)(ws + OFF_BAR);

    const int tid  = threadIdx.x;
    const int g    = blockIdx.x;            // 0..15: hidden-unit group
    const int mt   = tid >> 5;              // wave id == M tile
    const int lane = tid & 31;
    const int n    = g * 16 + (lane & 15);  // hidden unit 0..255
    const int Mb   = mt * 16 + (lane >> 4) * 8;

    const float4 b0v = make_float4(bias0[n], bias0[n + 256], bias0[n + 512], bias0[n + 768]);
    const float4 b1v = make_float4(bias1[n], bias1[n + 256], bias1[n + 512], bias1[n + 768]);
    float h[8];

    // ---------- backward direction: single step on x[T-1], zero init ----------
    #pragma unroll
    for (int r = 0; r < 8; ++r) {
        int M = Mb + r;
        packA(Ap0, M, n, x[(size_t)M * T_SEQ * I_IN + (size_t)(T_SEQ - 1) * I_IN + n]);
        packA(Ap0, M, HID + n, 0.0f);
    }
    grid_sync(bar, bar + 1);
    gemm_act(Ap0, Bp0, ldsB, g, mt, lane, n, b0v, cb, true, h);
    #pragma unroll
    for (int r = 0; r < 8; ++r) {
        int M = Mb + r;
        packA(Ap1, M, n, h[r]);
        packA(Ap1, M, HID + n, 0.0f);
    }
    grid_sync(bar, bar + 1);
    gemm_act(Ap1, Bp1, ldsB, g, mt, lane, n, b1v, cb, true, h);
    #pragma unroll
    for (int r = 0; r < 8; ++r) {
        int M = Mb + r;
        int idx = M * HID + n;
        hb1[idx] = h[r];
        h0[idx] = 0.0f; c0[idx] = 0.0f; h1[idx] = 0.0f; c1[idx] = 0.0f;
        packA(Ap0, M, n, x[(size_t)M * T_SEQ * I_IN + n]);   // Apack0(t=0) = [x_0 | 0]
        packA(Ap0, M, HID + n, 0.0f);
    }
    grid_sync(bar, bar + 1);

    // ---------- forward direction: T sequential steps, 2 grid syncs each ----------
    for (int t = 0; t < T_SEQ; ++t) {
        // layer 0: gates+update fused in registers
        gemm_act(Ap0, Bp0, ldsB, g, mt, lane, n, b0v, c0, false, h);
        #pragma unroll
        for (int r = 0; r < 8; ++r) {
            int M = Mb + r;
            int idx = M * HID + n;
            h0[idx] = h[r];
            packA(Ap1, M, n, h[r]);                 // Apack1 = [h0_new | h1_old]
            packA(Ap1, M, HID + n, h1[idx]);
            if (t + 1 < T_SEQ)                      // warm x[t+1] while layer1 runs
                __builtin_prefetch((const void*)&x[(size_t)M * T_SEQ * I_IN +
                                                   (size_t)(t + 1) * I_IN + n], 0, 0);
        }
        grid_sync(bar, bar + 1);

        // layer 1
        gemm_act(Ap1, Bp1, ldsB, g, mt, lane, n, b1v, c1, false, h);
        #pragma unroll
        for (int r = 0; r < 8; ++r) {
            int M = Mb + r;
            int idx = M * HID + n;
            h1[idx] = h[r];
            if (t + 1 < T_SEQ) {                    // Apack0 = [x_{t+1} | h0_new]
                packA(Ap0, M, n, x[(size_t)M * T_SEQ * I_IN + (size_t)(t + 1) * I_IN + n]);
                packA(Ap0, M, HID + n, h0[idx]);
            }
        }
        grid_sync(bar, bar + 1);
    }

    // ---------- final FC: out = [h1_fwd | h_bwd] @ fcW^T + fcb ----------
    #pragma unroll
    for (int r = 0; r < 8; ++r) {
        int M = Mb + r;
        int idx = M * HID + n;
        packA(Ap0, M, n, h1[idx]);
        packA(Ap0, M, HID + n, hb1[idx]);
    }
    grid_sync(bar, bar + 1);

    v8f acc = {};
    v16bf areg = *(const v16bf*)(Ap0 + ((size_t)(mt * 16) << 9) + lane * 16);
    #pragma unroll
    for (int kt = 0; kt < 16; ++kt) {
        const int ktn = (kt + 1 < 16) ? kt + 1 : kt;
        v16bf anext = *(const v16bf*)(Ap0 + ((size_t)(mt * 16 + ktn) << 9) + lane * 16);
        v16bf bb    = *(const v16bf*)(BpF + ((size_t)(kt * 16 + g) << 9) + lane * 16);
        acc = __builtin_amdgcn_wmma_f32_16x16x32_bf16(false, areg, false, bb, (short)0, acc, false, false);
        areg = anext;
    }
    const float bias = fcb[n];
    #pragma unroll
    for (int r = 0; r < 8; ++r)
        out[(size_t)(Mb + r) * 256 + n] = acc[r] + bias;
}

extern "C" void kernel_launch(void* const* d_in, const int* in_sizes, int n_in,
                              void* d_out, int out_size, void* d_ws, size_t ws_size,
                              hipStream_t stream)
{
    (void)in_sizes; (void)n_in; (void)out_size; (void)ws_size;
    const float* x   = (const float*)d_in[0];
    const float* Wxh = (const float*)d_in[1];
    const float* Whh = (const float*)d_in[2];
    const float* bxh = (const float*)d_in[3];
    const float* bhh = (const float*)d_in[4];
    const float* fcW = (const float*)d_in[5];
    const float* fcb = (const float*)d_in[6];
    unsigned char* ws = (unsigned char*)d_ws;
    float* out = (float*)d_out;

    pack_gate_weights<<<256, 256, 0, stream>>>(Wxh, Whh, (__bf16*)(ws + OFF_BPACK0));
    pack_fc_weights<<<32, 256, 0, stream>>>(fcW, (__bf16*)(ws + OFF_BPACKF));
    pack_bias<<<8, 256, 0, stream>>>(bxh, bhh, (float*)(ws + OFF_BIAS0),
                                     (unsigned*)(ws + OFF_BAR));
    lstm_persistent<<<NWG, NTHR, 0, stream>>>(x, fcb, ws, out);
}
